// BiaffineNER_40329742909578
// MI455X (gfx1250) — compile-verified, matrix operand order checked by reading, not run on hardware
//
#include <hip/hip_runtime.h>
#include <hip/hip_bf16.h>

typedef __attribute__((ext_vector_type(16))) __bf16 v16bf;
typedef __attribute__((ext_vector_type(8)))  __bf16 v8bf;
typedef __attribute__((ext_vector_type(8)))  float  v8f;

// ---------------- problem constants ----------------
#define BB   64
#define TT   512
#define DD   768
#define HH   256
#define LL   9
#define MTOT (BB*TT)      // 32768
#define FH   (4*HH)       // 1024
#define H2   (2*HH)       // 512

// ---------------- WMMA helpers ----------------
__device__ __forceinline__ v8f wmma_bf16(v16bf a, v16bf b, v8f c) {
    return __builtin_amdgcn_wmma_f32_16x16x32_bf16(false, a, false, b, (short)0, c, false, false);
}

// 16-bit fragment layout (ISA 7.12.2): lane<16 -> rows M=lane, elems 0..7 = K[k0..k0+7],
// elems 8..15 = K[k0+16..k0+23]; lanes 16..31 shift K by +8.
__device__ __forceinline__ v16bf frag_from_bf16(const __bf16* p) {
    v8bf lo = *(const v8bf*)p;
    v8bf hi = *(const v8bf*)(p + 16);
    v16bf r;
#pragma unroll
    for (int j = 0; j < 8; ++j) { r[j] = lo[j]; r[j + 8] = hi[j]; }
    return r;
}

// B fragment from row-major bf16 weight [N, ldk] (i.e. B(k,n) = W[n*ldk+k])
__device__ __forceinline__ v16bf load_bfrag(const __bf16* W, int ldk, int n0, int k0, int lane) {
    return frag_from_bf16(W + (size_t)(n0 + (lane & 15)) * ldk + k0 + ((lane >> 4) << 3));
}

// A fragment from an f32 tile (LDS or global), converted to bf16 in-register
__device__ __forceinline__ v16bf load_afrag_f32(const float* S, int ldk, int m0, int k0, int lane) {
    const float* p = S + (size_t)(m0 + (lane & 15)) * ldk + k0 + ((lane >> 4) << 3);
    v16bf r;
#pragma unroll
    for (int j = 0; j < 8; ++j) { r[j] = (__bf16)p[j]; r[j + 8] = (__bf16)p[16 + j]; }
    return r;
}

// A fragment from LDS bf16 [rows, ldk]
__device__ __forceinline__ v16bf load_afrag_lds(const __bf16* S, int ldk, int m0, int k0, int lane) {
    return frag_from_bf16(S + (size_t)(m0 + (lane & 15)) * ldk + k0 + ((lane >> 4) << 3));
}

// async global->LDS copy of 16 bytes (ASYNCcnt path); lds offset is allocation-relative
__device__ __forceinline__ void async_g2l_b128(unsigned lds_off, const float* gsrc) {
    asm volatile("global_load_async_to_lds_b128 %0, %1, off"
                 :: "v"(lds_off), "v"(gsrc) : "memory");
}
__device__ __forceinline__ void wait_async0() {
    asm volatile("s_wait_asynccnt 0x0" ::: "memory");
}

// ---------------- fp32 -> bf16 convert ----------------
__global__ void f2bf_kernel(const float* __restrict__ in, __bf16* __restrict__ out, int n) {
    int i = blockIdx.x * 256 + threadIdx.x;
    if (i < n) out[i] = (__bf16)in[i];
}

// ---------------- big GEMM: C[M,N] = A[M,K](f32) * W[N,K]^T(bf16) + bias[N] ----------------
// block = 256 thr (8 waves, 4x2), block tile 128x128, wave tile 32x64 (2x4 WMMA frags).
// A k-slab (128x32 f32, 16KB) double-buffered in LDS via async global->LDS copies.
__global__ __launch_bounds__(256)
void gemm_bias_kernel(const float* __restrict__ A, const __bf16* __restrict__ W,
                      const float* __restrict__ bias, float* __restrict__ C,
                      int M, int N, int K) {
    __shared__ float Ast[2][128 * 32];     // 2 x 16KB double buffer
    int tid = threadIdx.x;
    int w = tid >> 5, lane = tid & 31;
    int bm = blockIdx.y * 128;
    int m0w = (w & 3) * 32;                            // block-relative wave rows
    int n0 = blockIdx.x * 128 + (w >> 2) * 64;         // absolute wave cols

    unsigned stage_base[2];
    stage_base[0] = (unsigned)(size_t)(&Ast[0][0]);
    stage_base[1] = (unsigned)(size_t)(&Ast[1][0]);

    // issue async copy of A[bm..bm+128, kb..kb+32) into buffer `buf`
    auto issue_tile = [&](int buf, int kb) {
#pragma unroll
        for (int i = 0; i < 4; ++i) {
            int chunk = tid + i * 256;                 // 1024 x 16B chunks
            int row = chunk >> 3, c = chunk & 7;
            const float* gsrc = A + (size_t)(bm + row) * K + kb + c * 4;
            unsigned loff = stage_base[buf] + (unsigned)((row * 32 + c * 4) << 2);
            async_g2l_b128(loff, gsrc);
        }
    };

    v8f acc[2][4] = {};
    issue_tile(0, 0);
    int cur = 0;
    for (int kb = 0; kb < K; kb += 32) {
        wait_async0();          // our async writes into Ast[cur] done
        __syncthreads();        // everyone's done (also: prev buffer fully consumed)
        if (kb + 32 < K) issue_tile(cur ^ 1, kb + 32);

        v16bf a0 = load_afrag_f32(&Ast[cur][0], 32, m0w, 0, lane);
        v16bf a1 = load_afrag_f32(&Ast[cur][0], 32, m0w + 16, 0, lane);
        v16bf b0 = load_bfrag(W, K, n0, kb, lane);
        v16bf b1 = load_bfrag(W, K, n0 + 16, kb, lane);
        v16bf b2 = load_bfrag(W, K, n0 + 32, kb, lane);
        v16bf b3 = load_bfrag(W, K, n0 + 48, kb, lane);
        acc[0][0] = wmma_bf16(a0, b0, acc[0][0]);
        acc[0][1] = wmma_bf16(a0, b1, acc[0][1]);
        acc[0][2] = wmma_bf16(a0, b2, acc[0][2]);
        acc[0][3] = wmma_bf16(a0, b3, acc[0][3]);
        acc[1][0] = wmma_bf16(a1, b0, acc[1][0]);
        acc[1][1] = wmma_bf16(a1, b1, acc[1][1]);
        acc[1][2] = wmma_bf16(a1, b2, acc[1][2]);
        acc[1][3] = wmma_bf16(a1, b3, acc[1][3]);
        cur ^= 1;
    }
    int lm = (lane >> 4) * 8, ln = lane & 15;
#pragma unroll
    for (int mi = 0; mi < 2; ++mi)
#pragma unroll
        for (int ni = 0; ni < 4; ++ni)
#pragma unroll
            for (int v = 0; v < 8; ++v) {
                int m = bm + m0w + mi * 16 + v + lm;
                int n = n0 + ni * 16 + ln;
                if (m < M) C[(size_t)m * N + n] = acc[mi][ni][v] + bias[n];
            }
}

// ---------------- persistent LSTM scan (one workgroup per direction) ----------------
// LDS: h bf16 [64][256] (32KB) + c f32 [64][256] (64KB) + raw gates bf16 [64][1024] (128KB)
__global__ __launch_bounds__(1024)
void lstm_scan_kernel(const float* __restrict__ xgF, const float* __restrict__ xgB,
                      const __bf16* __restrict__ WhF, const __bf16* __restrict__ WhB,
                      const float* __restrict__ bhF, const float* __restrict__ bhB,
                      float* __restrict__ hcat) {
    extern __shared__ char smem[];
    __bf16* hs = (__bf16*)smem;                                  // 64*256 bf16
    float*  cs = (float*)(smem + BB * HH * 2);                   // 64*256 f32
    __bf16* gs = (__bf16*)(smem + BB * HH * 2 + BB * HH * 4);    // 64*1024 bf16

    int dir = blockIdx.x;
    const float*  xg  = dir ? xgB : xgF;
    const __bf16* Whh = dir ? WhB : WhF;
    const float*  bhh = dir ? bhB : bhF;

    int tid = threadIdx.x, w = tid >> 5, lane = tid & 31;
    for (int i = tid; i < BB * HH; i += 1024) { hs[i] = (__bf16)0.f; cs[i] = 0.f; }
    __syncthreads();

    int n0 = w * 32;                       // each wave owns a 32-wide gate column strip
    int lm = (lane >> 4) * 8, ln = lane & 15;

    for (int s = 0; s < TT; ++s) {
        int t = dir ? (TT - 1 - s) : s;
        // ---- GEMM: raw = h[64,256] @ Whh^T -> [64, n0..n0+32) ----
        v8f acc[4][2] = {};
        for (int kb = 0; kb < HH; kb += 32) {
            v16bf b0 = load_bfrag(Whh, HH, n0, kb, lane);
            v16bf b1 = load_bfrag(Whh, HH, n0 + 16, kb, lane);
#pragma unroll
            for (int mt = 0; mt < 4; ++mt) {
                v16bf a = load_afrag_lds(hs, HH, mt * 16, kb, lane);
                acc[mt][0] = wmma_bf16(a, b0, acc[mt][0]);
                acc[mt][1] = wmma_bf16(a, b1, acc[mt][1]);
            }
        }
#pragma unroll
        for (int mt = 0; mt < 4; ++mt)
#pragma unroll
            for (int ni = 0; ni < 2; ++ni)
#pragma unroll
                for (int v = 0; v < 8; ++v)
                    gs[(size_t)(mt * 16 + v + lm) * FH + n0 + ni * 16 + ln] = (__bf16)acc[mt][ni][v];
        __syncthreads();   // gates complete; all hs reads done

        // ---- gate nonlinearities + state update ----
        for (int idx = tid; idx < BB * HH; idx += 1024) {
            int m = idx >> 8, j = idx & 255;
            const float* xr = xg + ((size_t)m * TT + t) * FH;
            float gi = (float)gs[(size_t)m * FH + j]           + xr[j]           + bhh[j];
            float gf = (float)gs[(size_t)m * FH + HH + j]      + xr[HH + j]      + bhh[HH + j];
            float gg = (float)gs[(size_t)m * FH + 2 * HH + j]  + xr[2 * HH + j]  + bhh[2 * HH + j];
            float go = (float)gs[(size_t)m * FH + 3 * HH + j]  + xr[3 * HH + j]  + bhh[3 * HH + j];
            float i_ = 1.f / (1.f + __expf(-gi));
            float f_ = 1.f / (1.f + __expf(-gf));
            float o_ = 1.f / (1.f + __expf(-go));
            float g_ = tanhf(gg);
            float c  = f_ * cs[idx] + i_ * g_;
            cs[idx]  = c;
            float h  = o_ * tanhf(c);
            hs[idx]  = (__bf16)h;
            hcat[((size_t)m * TT + t) * H2 + dir * HH + j] = h;
            if (s + 1 < TT) {  // pull next timestep's gate slab toward the WGP
                int tn = dir ? (TT - 2 - s) : (s + 1);
                __builtin_prefetch(xg + ((size_t)m * TT + tn) * FH + j, 0, 1);
            }
        }
        __syncthreads();   // new h visible before next step's GEMM
    }
}

// ---------------- classifier: em[mt,l] = h[mt,:] . W[l,:] + b[l] ----------------
__global__ void cls_kernel(const float* __restrict__ hc, const float* __restrict__ Wc,
                           const float* __restrict__ bc, float* __restrict__ em) {
    int gid = blockIdx.x * 256 + threadIdx.x;
    if (gid >= MTOT * LL) return;
    int mt = gid / LL, l = gid % LL;
    const float* hr = hc + (size_t)mt * H2;
    const float* wr = Wc + (size_t)l * H2;
    float s = bc[l];
    for (int k = 0; k < H2; ++k) s += hr[k] * wr[k];
    em[gid] = s;
}

// ---------------- CRF NLL scan: one wave per batch element ----------------
__global__ __launch_bounds__(32)
void crf_kernel(const float* __restrict__ em, const int* __restrict__ labels,
                const float* __restrict__ st, const float* __restrict__ en,
                const float* __restrict__ tr, float* __restrict__ part) {
    int b = blockIdx.x, lane = threadIdx.x;
    __shared__ float alpha[16];
    float trc[LL];
    if (lane < LL)
        for (int i = 0; i < LL; ++i) trc[i] = tr[i * LL + lane];
    const float* eb = em + (size_t)b * TT * LL;
    const int*   lb = labels + (size_t)b * TT;
    float score = 0.f; int prev = 0;
    if (lane < LL) alpha[lane] = st[lane] + eb[lane];
    if (lane == 0) { prev = lb[0]; score = st[prev] + eb[prev]; }
    __syncthreads();
    for (int t = 1; t < TT; ++t) {
        const float* et = eb + t * LL;
        bool mt = lb[t] > -1;
        float nxt = 0.f;
        if (lane < LL) {
            float mx = -3.4e38f;
            for (int i = 0; i < LL; ++i) mx = fmaxf(mx, alpha[i] + trc[i]);
            float s = 0.f;
            for (int i = 0; i < LL; ++i) s += __expf(alpha[i] + trc[i] - mx);
            nxt = mx + __logf(s) + et[lane];
        }
        if (lane == 0 && mt) {
            int tg = lb[t];
            score += tr[prev * LL + tg] + et[tg];
            prev = tg;
        }
        __syncthreads();
        if (lane < LL && mt) alpha[lane] = nxt;
        __syncthreads();
    }
    if (lane == 0) {
        score += en[prev];
        float mx = -3.4e38f;
        for (int j = 0; j < LL; ++j) mx = fmaxf(mx, alpha[j] + en[j]);
        float s = 0.f;
        for (int j = 0; j < LL; ++j) s += __expf(alpha[j] + en[j] - mx);
        part[b] = (mx + __logf(s)) - score;
    }
}

__global__ void reduce_kernel(const float* __restrict__ part, float* __restrict__ out) {
    if (threadIdx.x == 0) {
        float a = 0.f;
        for (int k = 0; k < BB; ++k) a += part[k];
        out[0] = a;
    }
}

// ---------------- host orchestration ----------------
extern "C" void kernel_launch(void* const* d_in, const int* in_sizes, int n_in,
                              void* d_out, int out_size, void* d_ws, size_t ws_size,
                              hipStream_t stream) {
    const float* x       = (const float*)d_in[0];
    const int*   labels  = (const int*)d_in[1];
    const float* Wif0 = (const float*)d_in[2],  *Whf0 = (const float*)d_in[3];
    const float* bif0 = (const float*)d_in[4],  *bhf0 = (const float*)d_in[5];
    const float* Wib0 = (const float*)d_in[6],  *Whb0 = (const float*)d_in[7];
    const float* bib0 = (const float*)d_in[8],  *bhb0 = (const float*)d_in[9];
    const float* Wif1 = (const float*)d_in[10], *Whf1 = (const float*)d_in[11];
    const float* bif1 = (const float*)d_in[12], *bhf1 = (const float*)d_in[13];
    const float* Wib1 = (const float*)d_in[14], *Whb1 = (const float*)d_in[15];
    const float* bib1 = (const float*)d_in[16], *bhb1 = (const float*)d_in[17];
    const float* clsW = (const float*)d_in[18], *clsB = (const float*)d_in[19];
    const float* crfS = (const float*)d_in[20], *crfE = (const float*)d_in[21];
    const float* crfT = (const float*)d_in[22];

    char* ws = (char*)d_ws;
    // bf16 weight block (element offsets)
    __bf16* wbf = (__bf16*)ws;
    const size_t oWiF0 = 0,        oWhF0 = 786432,  oWiB0 = 1048576, oWhB0 = 1835008;
    const size_t oWiF1 = 2097152,  oWhF1 = 2621440, oWiB1 = 2883584, oWhB1 = 3407872;
    float* xg_f  = (float*)(ws + 7340032);              // 32768*1024 f32
    float* xg_b  = (float*)(ws + 7340032 + 134217728ull);
    float* hcat0 = (float*)(ws + 7340032 + 2ull * 134217728ull);
    float* hcat1 = (float*)(ws + 7340032 + 2ull * 134217728ull + 67108864ull);
    float* em    = (float*)(ws + 7340032 + 2ull * 134217728ull + 2ull * 67108864ull);
    float* part  = (float*)(ws + 7340032 + 2ull * 134217728ull + 2ull * 67108864ull + 1179648ull);
    (void)ws_size; (void)n_in; (void)in_sizes; (void)out_size;

    auto cvt = [&](const float* src, __bf16* dst, int n) {
        f2bf_kernel<<<(n + 255) / 256, 256, 0, stream>>>(src, dst, n);
    };
    cvt(Wif0, wbf + oWiF0, FH * DD);
    cvt(Whf0, wbf + oWhF0, FH * HH);
    cvt(Wib0, wbf + oWiB0, FH * DD);
    cvt(Whb0, wbf + oWhB0, FH * HH);
    cvt(Wif1, wbf + oWiF1, FH * H2);
    cvt(Whf1, wbf + oWhF1, FH * HH);
    cvt(Wib1, wbf + oWiB1, FH * H2);
    cvt(Whb1, wbf + oWhB1, FH * HH);

    dim3 gg(FH / 128, MTOT / 128);
    // layer 0 input projections
    gemm_bias_kernel<<<gg, 256, 0, stream>>>(x, wbf + oWiF0, bif0, xg_f, MTOT, FH, DD);
    gemm_bias_kernel<<<gg, 256, 0, stream>>>(x, wbf + oWiB0, bib0, xg_b, MTOT, FH, DD);
    // layer 0 recurrence (fwd+bwd concurrently)
    size_t lds = (size_t)BB * HH * 2 + (size_t)BB * HH * 4 + (size_t)BB * FH * 2; // 224 KB
    lstm_scan_kernel<<<2, 1024, lds, stream>>>(xg_f, xg_b, wbf + oWhF0, wbf + oWhB0,
                                               bhf0, bhb0, hcat0);
    // layer 1 input projections from concatenated hidden states
    gemm_bias_kernel<<<gg, 256, 0, stream>>>(hcat0, wbf + oWiF1, bif1, xg_f, MTOT, FH, H2);
    gemm_bias_kernel<<<gg, 256, 0, stream>>>(hcat0, wbf + oWiB1, bib1, xg_b, MTOT, FH, H2);
    lstm_scan_kernel<<<2, 1024, lds, stream>>>(xg_f, xg_b, wbf + oWhF1, wbf + oWhB1,
                                               bhf1, bhb1, hcat1);
    // classifier + CRF
    cls_kernel<<<(MTOT * LL + 255) / 256, 256, 0, stream>>>(hcat1, clsW, clsB, em);
    crf_kernel<<<BB, 32, 0, stream>>>(em, labels, crfS, crfE, crfT, part);
    reduce_kernel<<<1, 32, 0, stream>>>(part, (float*)d_out);
}
